// SparseColumnAttention_223338300231
// MI455X (gfx1250) — matrix operand order repaired and checked by reading, not compiled, not run on hardware
//
#include <hip/hip_runtime.h>

typedef __attribute__((ext_vector_type(16))) _Float16 v16h;
typedef __attribute__((ext_vector_type(8)))  _Float16 v8h;
typedef __attribute__((ext_vector_type(4)))  _Float16 v4h;
typedef __attribute__((ext_vector_type(2)))  _Float16 v2h;
typedef __attribute__((ext_vector_type(8)))  float    v8f;
typedef __attribute__((ext_vector_type(4)))  float    v4f;

#define C_DIM   512
#define QKV_C   1536      // fused q|k|v output width
#define R_ROWS  11520     // 2 * 128 * 45
#define T_PRS   45
#define K_NEI   17
#define N_HEADS 8
#define D_HEAD  64

// ---------------------------------------------------------------------------
// f32 -> f16 conversion (vectorized by 4)
// ---------------------------------------------------------------------------
__global__ __launch_bounds__(256)
void cvt_f32_to_f16_kernel(const float* __restrict__ src,
                           _Float16* __restrict__ dst, int n4) {
    int i = blockIdx.x * blockDim.x + threadIdx.x;
    if (i >= n4) return;
    v4f s = *(const v4f*)(src + (size_t)i * 4);
    v4h h;
    h[0] = (_Float16)s.x; h[1] = (_Float16)s.y;
    h[2] = (_Float16)s.z; h[3] = (_Float16)s.w;
    *(v4h*)(dst + (size_t)i * 4) = h;
}

// ---------------------------------------------------------------------------
// A-fragment loader: 16-bit 16x32 A tile, this code's own k order. lane (r = lane&15, h = lane>>4)
// holds row r, K range [k0+16h, +16) (contiguous, matches B).
// ---------------------------------------------------------------------------
__device__ __forceinline__ v16h load_a_frag(const _Float16* __restrict__ aRow,
                                            int k0, int h) {
    v8h lo = *(const v8h*)(aRow + k0 + h * 16);
    v8h hi = *(const v8h*)(aRow + k0 + 8 + h * 16);
    v16h a;
#pragma unroll
    for (int i = 0; i < 8; ++i) { a[i] = lo[i]; a[8 + i] = hi[i]; }
    return a;
}

// ---------------------------------------------------------------------------
// Y[r,o] = sum_c A[r,c] * W[o,c] + bias[o]
//   A: [R, 512] f16 row-major.  W: [Cout, 512] f16 row-major.  Y: [R, Cout] f32.
// Grid: (R/32, Cout/512). Block: 256 threads = 8 waves; block owns 32 rows,
// wave w owns 64 output cols -> per wave: 2 A frags x 4 B frags -> 8 WMMAs
// per 32-wide K-step (0.75 KB of fragment loads per WMMA).
// K-loop software-pipelined (register double buffering) so the b128 loads for
// step k+1 overlap the 8 WMMAs of step k.
// ---------------------------------------------------------------------------
__global__ __launch_bounds__(256)
void gemm_wmma_kernel(const _Float16* __restrict__ A,
                      const _Float16* __restrict__ W,
                      const float* __restrict__ bias,
                      float* __restrict__ Y,
                      int R, int Cout) {
    const int lane = threadIdx.x & 31;
    const int wave = threadIdx.x >> 5;               // 0..7
    const int row0 = blockIdx.x * 32;
    const int o0   = blockIdx.y * 512 + wave * 64;
    const int r    = lane & 15;                      // A row in tile / B column
    const int h    = lane >> 4;                      // half-wave select
    if (row0 >= R) return;

    const _Float16* aRow[2];
    aRow[0] = A + (size_t)(row0 + r) * C_DIM;
    aRow[1] = A + (size_t)(row0 + 16 + r) * C_DIM;

    const _Float16* wRow[4];
#pragma unroll
    for (int nt = 0; nt < 4; ++nt)
        wRow[nt] = W + (size_t)(o0 + nt * 16 + r) * C_DIM + h * 16;

    v8f acc[2][4];
#pragma unroll
    for (int m = 0; m < 2; ++m)
#pragma unroll
        for (int nt = 0; nt < 4; ++nt)
#pragma unroll
            for (int i = 0; i < 8; ++i) acc[m][nt][i] = 0.0f;

    // prologue: fragments for k0 = 0
    v16h aCur[2];
#pragma unroll
    for (int m = 0; m < 2; ++m) aCur[m] = load_a_frag(aRow[m], 0, h);
    v16h bCur[4];
#pragma unroll
    for (int nt = 0; nt < 4; ++nt) bCur[nt] = *(const v16h*)(wRow[nt]);

#pragma unroll 2
    for (int k0 = 32; k0 < C_DIM; k0 += 32) {
        // issue next step's loads first (independent of current WMMAs)
        v16h aNxt[2];
#pragma unroll
        for (int m = 0; m < 2; ++m) aNxt[m] = load_a_frag(aRow[m], k0, h);
        v16h bNxt[4];
#pragma unroll
        for (int nt = 0; nt < 4; ++nt) bNxt[nt] = *(const v16h*)(wRow[nt] + k0);

        // 8 WMMAs; same B fragment used by two consecutive WMMAs
#pragma unroll
        for (int nt = 0; nt < 4; ++nt)
#pragma unroll
            for (int m = 0; m < 2; ++m)
                acc[m][nt] = __builtin_amdgcn_wmma_f32_16x16x32_f16(
                    false, aCur[m], false, bCur[nt], (short)0, acc[m][nt],
                    false, false);

#pragma unroll
        for (int m = 0; m < 2; ++m) aCur[m] = aNxt[m];
#pragma unroll
        for (int nt = 0; nt < 4; ++nt) bCur[nt] = bNxt[nt];
    }
    // epilogue
#pragma unroll
    for (int nt = 0; nt < 4; ++nt)
#pragma unroll
        for (int m = 0; m < 2; ++m)
            acc[m][nt] = __builtin_amdgcn_wmma_f32_16x16x32_f16(
                false, aCur[m], false, bCur[nt], (short)0, acc[m][nt],
                false, false);

    // D layout: VGPR i, lane -> row = row0 + m*16 + 8h + i,
    //                           col = o0 + nt*16 + (lane&15)
#pragma unroll
    for (int nt = 0; nt < 4; ++nt) {
        const int col = o0 + nt * 16 + r;
        const float b = bias[col];
#pragma unroll
        for (int m = 0; m < 2; ++m)
#pragma unroll
            for (int i = 0; i < 8; ++i) {
                const int row = row0 + m * 16 + h * 8 + i;
                Y[(size_t)row * Cout + col] = acc[m][nt][i] + b;
            }
    }
}

// ---------------------------------------------------------------------------
// Attention over fused QKV buffer [R, 1536] (q|k|v column slabs).
// One wave32 per (row, head); each lane owns 2 of the 64 head dims.
// ---------------------------------------------------------------------------
__global__ __launch_bounds__(256)
void attention_kernel(const float* __restrict__ qkv,
                      const int* __restrict__ pair_idxs,   // [45,17]
                      _Float16* __restrict__ atth) {
    const int waveG = blockIdx.x * (blockDim.x >> 5) + (threadIdx.x >> 5);
    const int lane  = threadIdx.x & 31;
    const int rowId = waveG >> 3;       // / N_HEADS
    const int head  = waveG & (N_HEADS - 1);
    if (rowId >= R_ROWS) return;

    const int t    = rowId % T_PRS;
    const int base = rowId - t;         // row of t==0 in this (b,m) slab
    const int col  = head * D_HEAD + lane * 2;

    const float* qp = qkv + (size_t)rowId * QKV_C + col;
    const float q0 = qp[0], q1 = qp[1];

    int   nr[K_NEI];
    float sc[K_NEI];
#pragma unroll
    for (int j = 0; j < K_NEI; ++j) {
        nr[j] = base + pair_idxs[t * K_NEI + j];
        const float* kp = qkv + (size_t)nr[j] * QKV_C + 512 + col;
        float p = q0 * kp[0] + q1 * kp[1];
#pragma unroll
        for (int off = 16; off >= 1; off >>= 1) p += __shfl_xor(p, off, 32);
        sc[j] = p * 0.125f;             // 1/sqrt(64)
    }

    float mx = sc[0];
#pragma unroll
    for (int j = 1; j < K_NEI; ++j) mx = fmaxf(mx, sc[j]);
    float sum = 0.0f;
#pragma unroll
    for (int j = 0; j < K_NEI; ++j) { sc[j] = __expf(sc[j] - mx); sum += sc[j]; }
    const float inv = 1.0f / sum;

    float a0 = 0.0f, a1 = 0.0f;
#pragma unroll
    for (int j = 0; j < K_NEI; ++j) {
        const float w = sc[j] * inv;
        const float* vp = qkv + (size_t)nr[j] * QKV_C + 1024 + col;
        a0 += w * vp[0];
        a1 += w * vp[1];
    }
    v2h o; o[0] = (_Float16)a0; o[1] = (_Float16)a1;
    *(v2h*)(atth + (size_t)rowId * C_DIM + col) = o;
}

// ---------------------------------------------------------------------------
extern "C" void kernel_launch(void* const* d_in, const int* in_sizes, int n_in,
                              void* d_out, int out_size, void* d_ws, size_t ws_size,
                              hipStream_t stream) {
    const float* x    = (const float*)d_in[0];
    const float* Wq   = (const float*)d_in[1];
    const float* bq   = (const float*)d_in[2];
    const float* Wk   = (const float*)d_in[3];
    const float* bk   = (const float*)d_in[4];
    const float* Wv   = (const float*)d_in[5];
    const float* bv   = (const float*)d_in[6];
    const float* Wp   = (const float*)d_in[7];
    const float* bp   = (const float*)d_in[8];
    const int*   pidx = (const int*)d_in[9];
    float* out = (float*)d_out;

    char* ws = (char*)d_ws;
    size_t off = 0;
    auto alloc = [&](size_t bytes) -> void* {
        void* p = ws + off;
        off += (bytes + 255) & ~(size_t)255;
        return p;
    };

    const size_t RC = (size_t)R_ROWS * C_DIM;
    const size_t WE = (size_t)C_DIM * C_DIM;

    _Float16* xh    = (_Float16*)alloc(RC * 2);
    _Float16* Wqkvh = (_Float16*)alloc(3 * WE * 2);   // Wq|Wk|Wv rows stacked
    _Float16* Wph   = (_Float16*)alloc(WE * 2);
    float*    bqkv  = (float*)alloc(QKV_C * 4);
    float*    qkv   = (float*)alloc((size_t)R_ROWS * QKV_C * 4);
    _Float16* atth  = (_Float16*)alloc(RC * 2);

    // fused bias vector bq|bk|bv (d2d async copies are graph-capture safe)
    hipMemcpyAsync(bqkv,        bq, C_DIM * 4, hipMemcpyDeviceToDevice, stream);
    hipMemcpyAsync(bqkv + 512,  bk, C_DIM * 4, hipMemcpyDeviceToDevice, stream);
    hipMemcpyAsync(bqkv + 1024, bv, C_DIM * 4, hipMemcpyDeviceToDevice, stream);

    // f32 -> f16 conversions (weights stacked into the fused QKV weight)
    {
        int n4 = (int)(RC / 4);
        cvt_f32_to_f16_kernel<<<(n4 + 255) / 256, 256, 0, stream>>>(x, xh, n4);
        int w4 = (int)(WE / 4);
        int wb = (w4 + 255) / 256;
        cvt_f32_to_f16_kernel<<<wb, 256, 0, stream>>>(Wq, Wqkvh,          w4);
        cvt_f32_to_f16_kernel<<<wb, 256, 0, stream>>>(Wk, Wqkvh + WE,     w4);
        cvt_f32_to_f16_kernel<<<wb, 256, 0, stream>>>(Wv, Wqkvh + 2 * WE, w4);
        cvt_f32_to_f16_kernel<<<wb, 256, 0, stream>>>(Wp, Wph,            w4);
    }

    // fused QKV projection: [R,512] x [1536,512]^T -> [R,1536]
    {
        dim3 grid(R_ROWS / 32, QKV_C / 512);   // (360, 3)
        gemm_wmma_kernel<<<grid, 256, 0, stream>>>(xh, Wqkvh, bqkv, qkv,
                                                   R_ROWS, QKV_C);
    }

    // one wave per (row, head): R*8 waves, 8 waves per block
    const int attBlocks = (R_ROWS * N_HEADS) / 8;  // 11520
    attention_kernel<<<attBlocks, 256, 0, stream>>>(qkv, pidx, atth);

    // output projection straight into d_out (f32)
    {
        dim3 grid(R_ROWS / 32, 1);
        gemm_wmma_kernel<<<grid, 256, 0, stream>>>(atth, Wph, bp, out,
                                                   R_ROWS, C_DIM);
    }
}
